// WRGN_70755291234537
// MI455X (gfx1250) — matrix operand order repaired and checked by reading, not compiled, not run on hardware
//
#include <hip/hip_runtime.h>
#include <hip/hip_bf16.h>

typedef _Float16 f16_t;
typedef __attribute__((ext_vector_type(16))) _Float16 v16h;
typedef __attribute__((ext_vector_type(8)))  float    v8f;

#define UNITS 32

__device__ __forceinline__ v8f wmma16x16x32(v16h a, v16h b, v8f c) {
  // D(16x16,f32) = A(16x32,f16) x B(32x16,f16) + C
  return __builtin_amdgcn_wmma_f32_16x16x32_f16(false, a, false, b, (short)0, c,
                                                false, false);
}

// ---- fast activations: use CDNA5 V_TANH_F32 when available ---------------
__device__ __forceinline__ float fast_tanh(float x) {
#if __has_builtin(__builtin_amdgcn_tanhf)
  return __builtin_amdgcn_tanhf(x);
#elif __has_builtin(__builtin_amdgcn_tanh_f32)
  return __builtin_amdgcn_tanh_f32(x);
#else
  // branch-free fallback: tanh(x) = 1 - 2/(exp(2x)+1), v_exp + v_rcp
  float e = __expf(2.0f * x);
  return 1.0f - 2.0f * __builtin_amdgcn_rcpf(e + 1.0f);
#endif
}

__device__ __forceinline__ float fast_sigmoid(float x) {
  // exact identity: sigmoid(x) = 0.5*tanh(x/2) + 0.5  (no division)
  return 0.5f * fast_tanh(0.5f * x) + 0.5f;
}

// B-matrix fragment: B[k][n] = W[n_global][k] (row-major W, leading dim `ld`).
// Layout per ISA 05_wmma B striping: lanes 0-15 hold K={0..7,16..23} (2/VGPR),
// lanes 16-31 hold K={8..15,24..31}, column N = 16*g + (lane&15).
__device__ __forceinline__ v16h load_bfrag(const float* __restrict__ W, int ld,
                                           int g, int kOff, int lane) {
  int col = lane & 15;
  int kh  = lane >> 4;
  const float* wr = W + (size_t)(16 * g + col) * ld + kOff + kh * 8;
  v16h b;
#pragma unroll
  for (int v = 0; v < 8; ++v) {
    b[2 * v]     = (f16_t)wr[2 * v + 8 * (v >> 2)];
    b[2 * v + 1] = (f16_t)wr[2 * v + 8 * (v >> 2) + 1];
  }
  return b;
}

// A-fragment K offset for 16-bit 16x32 A layout:
// lanes 0-15: K = {0..7, 16..23}; lanes 16-31: K = {8..15, 24..31}; 2 per VGPR.
__device__ __forceinline__ int a_k0(int v, int hi) {
  return ((v >> 2) << 4) + (hi << 3) + ((v & 3) << 1);
}

// ---------------------------------------------------------------------------
// GRU over tiles of 16 sequences + per-timestep scatter-add into hk.
// ---------------------------------------------------------------------------
__global__ __launch_bounds__(256) void gru_scatter_kernel(
    const float* __restrict__ h, const int* __restrict__ idx, int nSeq, int L,
    const float* __restrict__ Wih, const float* __restrict__ Whh,
    const float* __restrict__ bih, const float* __restrict__ bhh,
    float* __restrict__ hk) {
  __shared__ float hbufAll[8][16 * 34];  // padded stride 34 -> conflict-free
  __shared__ int   idxAll[8][16];

  const int lane = threadIdx.x & 31;
  const int wave = threadIdx.x >> 5;
  float* hbuf = hbufAll[wave];
  int*   idxb = idxAll[wave];

  const int row  = lane & 15;
  const int hi   = lane >> 4;
  const int base = (blockIdx.x * 8 + wave) * 16;

  // 12 weight B-fragments held in VGPRs across the whole sequence loop.
  v16h wih[6], whh[6];
#pragma unroll
  for (int g = 0; g < 6; ++g) {
    wih[g] = load_bfrag(Wih, UNITS, g, 0, lane);
    whh[g] = load_bfrag(Whh, UNITS, g, 0, lane);
  }

  // Per-lane gate biases for the two 16-column tiles this lane owns.
  float brz[2][2], bnI[2], bnH[2];
#pragma unroll
  for (int ct = 0; ct < 2; ++ct) {
    int c = 16 * ct + row;
    brz[ct][0] = bih[c] + bhh[c];            // r-gate bias (merged)
    brz[ct][1] = bih[32 + c] + bhh[32 + c];  // z-gate bias (merged)
    bnI[ct]    = bih[64 + c];                // n-gate input bias
    bnH[ct]    = bhh[64 + c];                // n-gate hidden bias
  }

  v8f zero = {};
  v8f hD[2];     // h_prev in C/D layout (cols 0-15, 16-31)
  hD[0] = zero;
  hD[1] = zero;
  v16h hA = {};  // h_prev as A-fragment (valid for t>0)

  for (int t = 0; t < L; ++t) {
    // Gather index for this lane's sequence row.
    int s  = base + row;
    int sc = (s < nSeq) ? s : 0;
    int xi = idx[(size_t)sc * L + t];
    if (hi == 0) idxb[row] = xi;  // one writer per row; read after barrier

    // Build X A-fragment straight from global memory (f32 -> f16).
    const float* xr = h + (size_t)xi * UNITS;
    v16h xA;
#pragma unroll
    for (int v = 0; v < 8; ++v) {
      int k0 = a_k0(v, hi);
      xA[2 * v]     = (f16_t)xr[k0];
      xA[2 * v + 1] = (f16_t)xr[k0 + 1];
    }

    // gi = x @ Wih^T   (6 tiles of 16x16)
    v8f gi[6];
#pragma unroll
    for (int g = 0; g < 6; ++g) gi[g] = wmma16x16x32(xA, wih[g], zero);

    // gh: r/z parts accumulate directly into gi via WMMA C; n part kept apart.
    v8f ghn[2];
    ghn[0] = zero;
    ghn[1] = zero;
    if (t > 0) {
#pragma unroll
      for (int g = 0; g < 4; ++g) gi[g] = wmma16x16x32(hA, whh[g], gi[g]);
      ghn[0] = wmma16x16x32(hA, whh[4], zero);
      ghn[1] = wmma16x16x32(hA, whh[5], zero);
    }

    // Elementwise GRU update entirely in the C/D register layout.
#pragma unroll
    for (int ct = 0; ct < 2; ++ct) {
#pragma unroll
      for (int v = 0; v < 8; ++v) {
        float r = fast_sigmoid(gi[ct][v] + brz[ct][0]);
        float z = fast_sigmoid(gi[2 + ct][v] + brz[ct][1]);
        float n = fast_tanh(gi[4 + ct][v] + bnI[ct] + r * (ghn[ct][v] + bnH[ct]));
        hD[ct][v] = (1.0f - z) * n + z * hD[ct][v];
      }
    }

    __syncthreads();
    // Stage h into LDS (for the D->A transpose) + scatter-add outputs.
#pragma unroll
    for (int ct = 0; ct < 2; ++ct) {
#pragma unroll
      for (int v = 0; v < 8; ++v) {
        int   M   = v + 8 * hi;  // matrix row for VGPR v in D layout
        float val = hD[ct][v];
        hbuf[M * 34 + 16 * ct + row] = val;
        int sM = base + M;
        if (sM < nSeq)
          atomicAdd(hk + (size_t)idxb[M] * UNITS + 16 * ct + row, val);
      }
    }
    __syncthreads();

    // Rebuild h_prev as an f16 A-fragment for the next timestep.
#pragma unroll
    for (int v = 0; v < 8; ++v) {
      int k0 = a_k0(v, hi);
      hA[2 * v]     = (f16_t)hbuf[row * 34 + k0];
      hA[2 * v + 1] = (f16_t)hbuf[row * 34 + k0 + 1];
    }
  }
}

// ---------------------------------------------------------------------------
// out = tanh([h|hk2|hk3|hk4] @ W1^T + b1) @ W2^T + b2   (K=128 -> 4 WMMA chunks)
// ---------------------------------------------------------------------------
__global__ __launch_bounds__(256) void fc_kernel(
    const float* __restrict__ h, const float* __restrict__ hk,
    const float* __restrict__ W1, const float* __restrict__ b1,
    const float* __restrict__ W2, const float* __restrict__ b2,
    float* __restrict__ out, int n1) {
  __shared__ float bufAll[8][16 * 34];

  const int lane = threadIdx.x & 31;
  const int wave = threadIdx.x >> 5;
  float* buf = bufAll[wave];
  const int row  = lane & 15;
  const int hi   = lane >> 4;
  const int base = (blockIdx.x * 8 + wave) * 16;

  v16h bW1[4][2];
#pragma unroll
  for (int c = 0; c < 4; ++c)
#pragma unroll
    for (int g = 0; g < 2; ++g) bW1[c][g] = load_bfrag(W1, 128, g, 32 * c, lane);
  v16h bW2[2];
  bW2[0] = load_bfrag(W2, 32, 0, 0, lane);
  bW2[1] = load_bfrag(W2, 32, 1, 0, lane);

  float b1v[2] = {b1[row], b1[16 + row]};
  float b2v[2] = {b2[row], b2[16 + row]};

  int r0 = base + row;
  int rc = (r0 < n1) ? r0 : (n1 - 1);

  v8f zero = {};
  v8f d0 = zero, d1 = zero;
#pragma unroll
  for (int c = 0; c < 4; ++c) {
    const float* src = (c == 0) ? (h + (size_t)rc * UNITS)
                                : (hk + ((size_t)(c - 1) * n1 + rc) * UNITS);
    v16h a;
#pragma unroll
    for (int v = 0; v < 8; ++v) {
      int k0 = a_k0(v, hi);
      a[2 * v]     = (f16_t)src[k0];
      a[2 * v + 1] = (f16_t)src[k0 + 1];
    }
    d0 = wmma16x16x32(a, bW1[c][0], d0);
    d1 = wmma16x16x32(a, bW1[c][1], d1);
  }

  // bias + tanh, then D->A transpose through LDS.
#pragma unroll
  for (int v = 0; v < 8; ++v) {
    int M = v + 8 * hi;
    buf[M * 34 + row]      = fast_tanh(d0[v] + b1v[0]);
    buf[M * 34 + 16 + row] = fast_tanh(d1[v] + b1v[1]);
  }
  __syncthreads();

  v16h a2;
#pragma unroll
  for (int v = 0; v < 8; ++v) {
    int k0 = a_k0(v, hi);
    a2[2 * v]     = (f16_t)buf[row * 34 + k0];
    a2[2 * v + 1] = (f16_t)buf[row * 34 + k0 + 1];
  }
  v8f o0 = wmma16x16x32(a2, bW2[0], zero);
  v8f o1 = wmma16x16x32(a2, bW2[1], zero);

#pragma unroll
  for (int v = 0; v < 8; ++v) {
    int M  = v + 8 * hi;
    int rr = base + M;
    if (rr < n1) {
      out[(size_t)rr * UNITS + row]      = o0[v] + b2v[0];
      out[(size_t)rr * UNITS + 16 + row] = o1[v] + b2v[1];
    }
  }
}

extern "C" void kernel_launch(void* const* d_in, const int* in_sizes, int n_in,
                              void* d_out, int out_size, void* d_ws,
                              size_t ws_size, hipStream_t stream) {
  const float* h    = (const float*)d_in[0];
  const int*   idx2 = (const int*)d_in[1];
  const int*   idx3 = (const int*)d_in[2];
  const int*   idx4 = (const int*)d_in[3];
  const float* Wih  = (const float*)d_in[4];
  const float* Whh  = (const float*)d_in[5];
  const float* bih  = (const float*)d_in[6];
  const float* bhh  = (const float*)d_in[7];
  const float* W1   = (const float*)d_in[8];
  const float* b1   = (const float*)d_in[9];
  const float* W2   = (const float*)d_in[10];
  const float* b2   = (const float*)d_in[11];
  float* out = (float*)d_out;

  const int n1 = in_sizes[0] / UNITS;
  const int n2 = in_sizes[1] / 2;
  const int n3 = in_sizes[2] / 3;
  const int n4 = in_sizes[3] / 4;

  // Workspace: 3 accumulators of (n1 x 32) f32 = 76.8 MB. Capture-safe zeroing.
  float* hk = (float*)d_ws;
  hipMemsetAsync(d_ws, 0, (size_t)3 * n1 * UNITS * sizeof(float), stream);

  auto gru_blocks = [](int nSeq) {
    int tiles = (nSeq + 15) / 16;
    return (tiles + 7) / 8;  // 8 waves (tiles) per 256-thread block
  };
  gru_scatter_kernel<<<gru_blocks(n2), 256, 0, stream>>>(
      h, idx2, n2, 2, Wih, Whh, bih, bhh, hk);
  gru_scatter_kernel<<<gru_blocks(n3), 256, 0, stream>>>(
      h, idx3, n3, 3, Wih, Whh, bih, bhh, hk + (size_t)n1 * UNITS);
  gru_scatter_kernel<<<gru_blocks(n4), 256, 0, stream>>>(
      h, idx4, n4, 4, Wih, Whh, bih, bhh, hk + (size_t)2 * n1 * UNITS);

  int fcBlocks = ((n1 + 15) / 16 + 7) / 8;
  fc_kernel<<<fcBlocks, 256, 0, stream>>>(h, hk, W1, b1, W2, b2, out, n1);
}
